// SimpleSelfAttention_19181323944218
// MI455X (gfx1250) — compile-verified, matrix operand order checked by reading, not run on hardware
//
#include <hip/hip_runtime.h>
#include <hip/hip_bf16.h>
#include <math.h>

// ---------------------------------------------------------------------------
// CDNA5 (gfx1250) self-attention: bf16 WMMA GEMMs with async-to-LDS staging
// + flash-attention. wave32, v_wmma_f32_16x16x32_bf16 everywhere.
// ---------------------------------------------------------------------------

typedef __attribute__((ext_vector_type(16))) __bf16 v16bf;
typedef __attribute__((ext_vector_type(8)))  __bf16 v8bf;
typedef __attribute__((ext_vector_type(8)))  float  v8f;

#define SEQ   3072
#define DIMN  2048
#define NHEAD 16
#define HDIM  128
#define GF    8
#define GH    16
#define GW    24
#define LDSP  40   // padded LDS row stride (halves): 80B, 16B-aligned, bank-spread

union FragU { v16bf v; v8bf h[2]; };

// gfx1250 async global->LDS copy (ASYNCcnt path), with compile-safe fallback.
#if __has_builtin(__builtin_amdgcn_global_load_async_to_lds_b128) && \
    __has_builtin(__builtin_amdgcn_s_wait_asynccnt)
#define ATHENA_ASYNC_LDS 1
#endif

#ifdef ATHENA_ASYNC_LDS
// exact parameter types per hipcc diagnostic:
//   param0: 'int __attribute__((vector_size(16))) __device__ *'  (AS1 source)
//   param1: LDS destination (AS3), same element type
typedef int v4i_vs __attribute__((vector_size(16)));
typedef __attribute__((address_space(1))) v4i_vs* async_gptr;
typedef __attribute__((address_space(3))) v4i_vs* async_lptr;
#endif

__device__ inline void copy16_to_lds(const __bf16* g, __bf16* l) {
#ifdef ATHENA_ASYNC_LDS
  // AS1 ptr == generic global address; AS3 ptr == low 32 bits of generic LDS addr.
  __builtin_amdgcn_global_load_async_to_lds_b128(
      (async_gptr)(uintptr_t)g,
      (async_lptr)(unsigned int)(uintptr_t)l,
      0, 0);
#else
  *(v8bf*)l = *(const v8bf*)g;
#endif
}

__device__ inline void wait_lds_copies() {
#ifdef ATHENA_ASYNC_LDS
  __builtin_amdgcn_s_wait_asynccnt(0);
#endif
}

// Load a 16x32 bf16 A/B fragment. Per ISA 7.12.2 (16-bit A 16x32):
//   lanes 0-15 : M=lane,    halves 0..7 = K 0..7,  halves 8..15 = K 16..23
//   lanes 16-31: M=lane-16, halves 0..7 = K 8..15, halves 8..15 = K 24..31
__device__ inline v16bf load_frag(const __bf16* base, int row, int ld,
                                  int kbase, int lane) {
  int koff = (lane >> 4) * 8;
  const __bf16* p = base + (size_t)row * ld + kbase + koff;
  FragU f;
  f.h[0] = *(const v8bf*)(p);
  f.h[1] = *(const v8bf*)(p + 16);
  return f.v;
}

__device__ inline v16bf load_frag_lds(const __bf16* base, int row, int lane) {
  int koff = (lane >> 4) * 8;
  const __bf16* p = base + row * LDSP + koff;
  FragU f;
  f.h[0] = *(const v8bf*)(p);
  f.h[1] = *(const v8bf*)(p + 16);
  return f.v;
}

__device__ inline v8f wmma_bf16(v16bf a, v16bf b, v8f c) {
  return __builtin_amdgcn_wmma_f32_16x16x32_bf16(
      /*neg_a=*/false, a, /*neg_b=*/false, b,
      /*c_mod=*/(short)0, c, /*reuse_a=*/false, /*reuse_b=*/false);
}

// ---------------------------------------------------------------------------
// elementwise casts
// ---------------------------------------------------------------------------
__global__ void cast_f32_bf16(const float* __restrict__ in,
                              __bf16* __restrict__ out, int n) {
  int i = blockIdx.x * blockDim.x + threadIdx.x;
  if (i < n) out[i] = (__bf16)in[i];
}

// in: [rows, cols] f32 -> out: [cols, rows] bf16 (key index contiguous)
__global__ void transpose_cast_bf16(const float* __restrict__ in,
                                    __bf16* __restrict__ out,
                                    int rows, int cols) {
  int i = blockIdx.x * blockDim.x + threadIdx.x;
  if (i < rows * cols) {
    int r = i / cols, c = i - r * cols;
    out[(size_t)c * rows + r] = (__bf16)in[i];
  }
}

// ---------------------------------------------------------------------------
// GEMM: C[M,N] = A[M,K] (bf16) . B[N,K]^T (bf16) + bias[N]  (f32 out)
// block 256 thr = 8 waves; block tile 128x256; wave tile 64x64 (4x4 WMMA).
// K-slabs of 32 double-buffered in LDS via async global->LDS copies.
// ---------------------------------------------------------------------------
__global__ __launch_bounds__(256)
void gemm_bf16_nt(const __bf16* __restrict__ A, const __bf16* __restrict__ B,
                  const float* __restrict__ bias, float* __restrict__ C,
                  int M, int N, int K) {
  __shared__ __attribute__((aligned(16))) __bf16 sA[2][128 * LDSP];
  __shared__ __attribute__((aligned(16))) __bf16 sB[2][256 * LDSP];

  const int tid  = threadIdx.x;
  const int lane = tid & 31;
  const int wave = tid >> 5;
  const int wm   = wave >> 2;            // 0..1 (64 rows each)
  const int wn   = wave & 3;             // 0..3 (64 cols each)
  const int bm   = blockIdx.y * 128;
  const int bn   = blockIdx.x * 256;
  const int r    = lane & 15;

  // stage one 32-deep K slab of A (128x32) and B (256x32) into LDS buffer.
  // chunks are 16B (8 halves); a 32-half row = 4 chunks.
  auto stage = [&](int buf, int k) {
#pragma unroll
    for (int i = 0; i < 2; ++i) {              // A: 512 chunks / 256 threads
      int c = tid + i * 256;
      int row = c >> 2, sub = c & 3;
      copy16_to_lds(A + (size_t)(bm + row) * K + k + sub * 8,
                    &sA[buf][row * LDSP + sub * 8]);
    }
#pragma unroll
    for (int i = 0; i < 4; ++i) {              // B: 1024 chunks / 256 threads
      int c = tid + i * 256;
      int row = c >> 2, sub = c & 3;
      copy16_to_lds(B + (size_t)(bn + row) * K + k + sub * 8,
                    &sB[buf][row * LDSP + sub * 8]);
    }
  };

  v8f acc[4][4] = {};
  int buf = 0;
  stage(0, 0);

  for (int k = 0; k < K; k += 32) {
    wait_lds_copies();       // own wave's async copies landed
    __syncthreads();         // everyone's copies landed
    if (k + 32 < K) stage(buf ^ 1, k + 32);

    v16bf af[4], bfr[4];
#pragma unroll
    for (int mt = 0; mt < 4; ++mt)
      af[mt] = load_frag_lds(&sA[buf][0], wm * 64 + mt * 16 + r, lane);
#pragma unroll
    for (int nt = 0; nt < 4; ++nt)
      bfr[nt] = load_frag_lds(&sB[buf][0], wn * 64 + nt * 16 + r, lane);
#pragma unroll
    for (int mt = 0; mt < 4; ++mt)
#pragma unroll
      for (int nt = 0; nt < 4; ++nt)
        acc[mt][nt] = wmma_bf16(af[mt], bfr[nt], acc[mt][nt]);

    __syncthreads();         // all reads of buf done before it is re-staged
    buf ^= 1;
  }

  // C layout: VGPR j -> row j (lanes 0-15) / row j+8 (lanes 16-31); lane&15 = col
  const int rsel = (lane >> 4) * 8;
#pragma unroll
  for (int mt = 0; mt < 4; ++mt)
#pragma unroll
    for (int nt = 0; nt < 4; ++nt)
#pragma unroll
      for (int j = 0; j < 8; ++j) {
        int rr = bm + wm * 64 + mt * 16 + j + rsel;
        int cc = bn + wn * 64 + nt * 16 + (lane & 15);
        C[(size_t)rr * N + cc] = acc[mt][nt][j] + (bias ? bias[cc] : 0.0f);
      }
}

// ---------------------------------------------------------------------------
// fused RMSNorm (over 2048) + grid RoPE; writes bf16 (outscale folds 1/sqrt(d))
// one block (256 thr) per sequence row
// ---------------------------------------------------------------------------
__global__ __launch_bounds__(256)
void rmsnorm_rope(const float* __restrict__ in, const float* __restrict__ g,
                  const float* __restrict__ freqs /*[1024,64]*/,
                  __bf16* __restrict__ out, float outscale) {
  __shared__ float red[256];
  const int row = blockIdx.x;
  const int t   = threadIdx.x;
  const float* xr = in + (size_t)row * DIMN;

  float ss = 0.0f;
#pragma unroll
  for (int i = 0; i < 8; ++i) {
    float v = xr[t + i * 256];
    ss += v * v;
  }
  red[t] = ss;
  __syncthreads();
  for (int stt = 128; stt > 0; stt >>= 1) {
    if (t < stt) red[t] += red[t + stt];
    __syncthreads();
  }
  const float rinv = rsqrtf(red[0] * (1.0f / DIMN) + 1e-6f);

  // grid position -> angle source row
  const int f  = row / (GH * GW);
  const int hh = (row / GW) % GH;
  const int ww = row % GW;
  const int cf = 22, cfh = 43; // c=64: cf=22, ch=cw=21

#pragma unroll
  for (int p = 0; p < 4; ++p) {
    int col0 = t * 8 + 2 * p;
    int i    = (col0 & (HDIM - 1)) >> 1;               // pair idx in head, 0..63
    int sel  = (i < cf) ? f : ((i < cfh) ? hh : ww);
    float a  = freqs[sel * 64 + i];
    float ca = cosf(a), sa = sinf(a);
    float te = xr[col0]     * rinv * g[col0];
    float to = xr[col0 + 1] * rinv * g[col0 + 1];
    out[(size_t)row * DIMN + col0]     = (__bf16)((te * ca - to * sa) * outscale);
    out[(size_t)row * DIMN + col0 + 1] = (__bf16)((te * sa + to * ca) * outscale);
  }
}

// ---------------------------------------------------------------------------
// flash attention: grid (SEQ/64, NHEAD), block 128 (4 waves),
// each wave owns one 16-query tile of one head.
// Q,K: [SEQ, 2048] bf16 (scale folded into Q). Vt: [2048, SEQ] bf16.
// ---------------------------------------------------------------------------
__global__ __launch_bounds__(128)
void flash_attn(const __bf16* __restrict__ Q, const __bf16* __restrict__ Kb,
                const __bf16* __restrict__ Vt, __bf16* __restrict__ O) {
  __shared__ __attribute__((aligned(16))) __bf16 lds_p[4][16 * 32];
  const int lane = threadIdx.x & 31;
  const int wave = threadIdx.x >> 5;
  const int h    = blockIdx.y;
  const int qt   = blockIdx.x * 4 + wave;
  const int qrow = qt * 16 + (lane & 15);
  const int hoff = h * HDIM;
  const int rsel = (lane >> 4) * 8;
  __bf16* myp    = &lds_p[wave][0];

  // Q fragments for full head dim (4 x K=32)
  v16bf qa[4];
#pragma unroll
  for (int kk = 0; kk < 4; ++kk)
    qa[kk] = load_frag(Q, qrow, DIMN, hoff + kk * 32, lane);

  v8f   ot[8] = {};
  float mrow[8], lrow[8];
#pragma unroll
  for (int j = 0; j < 8; ++j) { mrow[j] = -1e30f; lrow[j] = 0.0f; }

  for (int kc = 0; kc < SEQ; kc += 32) {
    // S = Q . K^T for a 16x32 score tile
    v8f st[2] = {};
#pragma unroll
    for (int nt = 0; nt < 2; ++nt) {
      int krow = kc + nt * 16 + (lane & 15);
#pragma unroll
      for (int kk = 0; kk < 4; ++kk) {
        v16bf kf = load_frag(Kb, krow, DIMN, hoff + kk * 32, lane);
        st[nt] = wmma_bf16(qa[kk], kf, st[nt]);
      }
    }
    // online softmax per row; lanes 0-15 hold rows j, lanes 16-31 rows j+8,
    // so width-16 shuffles reduce each half independently.
#pragma unroll
    for (int j = 0; j < 8; ++j) {
      float v0 = st[0][j], v1 = st[1][j];
      float mx = fmaxf(v0, v1);
      mx = fmaxf(mx, __shfl_xor(mx, 1, 16));
      mx = fmaxf(mx, __shfl_xor(mx, 2, 16));
      mx = fmaxf(mx, __shfl_xor(mx, 4, 16));
      mx = fmaxf(mx, __shfl_xor(mx, 8, 16));
      float mnew  = fmaxf(mrow[j], mx);
      float alpha = __expf(mrow[j] - mnew);
      float p0 = __expf(v0 - mnew);
      float p1 = __expf(v1 - mnew);
      float rs = p0 + p1;
      rs += __shfl_xor(rs, 1, 16);
      rs += __shfl_xor(rs, 2, 16);
      rs += __shfl_xor(rs, 4, 16);
      rs += __shfl_xor(rs, 8, 16);
      lrow[j] = lrow[j] * alpha + rs;
      mrow[j] = mnew;
#pragma unroll
      for (int dt = 0; dt < 8; ++dt) ot[dt][j] = ot[dt][j] * alpha;
      // P tile to LDS (row-major 16x32 bf16) to re-form an A fragment
      int prow = j + rsel;
      myp[prow * 32 + (lane & 15)]      = (__bf16)p0;
      myp[prow * 32 + 16 + (lane & 15)] = (__bf16)p1;
    }
    // A fragment of P (DS ops are in-order within a wave)
    v16bf pa;
    {
      int rr = lane & 15, koff = (lane >> 4) * 8;
      FragU fr;
      fr.h[0] = *(const v8bf*)(&myp[rr * 32 + koff]);
      fr.h[1] = *(const v8bf*)(&myp[rr * 32 + koff + 16]);
      pa = fr.v;
    }
    // O += P . V  (V^T is d-major rows, key-contiguous)
#pragma unroll
    for (int dt = 0; dt < 8; ++dt) {
      v16bf vf = load_frag(Vt, hoff + dt * 16 + (lane & 15), SEQ, kc, lane);
      ot[dt] = wmma_bf16(pa, vf, ot[dt]);
    }
  }

  // normalize by l and emit bf16 [SEQ, 2048]
#pragma unroll
  for (int dt = 0; dt < 8; ++dt)
#pragma unroll
    for (int j = 0; j < 8; ++j) {
      int row = qt * 16 + j + rsel;
      int col = hoff + dt * 16 + (lane & 15);
      O[(size_t)row * DIMN + col] = (__bf16)(ot[dt][j] / lrow[j]);
    }
}

// ---------------------------------------------------------------------------
// launcher
// ---------------------------------------------------------------------------
extern "C" void kernel_launch(void* const* d_in, const int* in_sizes, int n_in,
                              void* d_out, int out_size, void* d_ws, size_t ws_size,
                              hipStream_t stream) {
  const float* x     = (const float*)d_in[0];
  const float* freqs = (const float*)d_in[1];
  const float* Wq    = (const float*)d_in[2];
  const float* bq    = (const float*)d_in[3];
  const float* Wk    = (const float*)d_in[4];
  const float* bk    = (const float*)d_in[5];
  const float* Wv    = (const float*)d_in[6];
  const float* bv    = (const float*)d_in[7];
  const float* Wo    = (const float*)d_in[8];
  const float* bo    = (const float*)d_in[9];
  const float* gq    = (const float*)d_in[10];
  const float* gk    = (const float*)d_in[11];
  float* outp        = (float*)d_out;

  // workspace carve-up (256B aligned)
  char*  base = (char*)d_ws;
  size_t off  = 0;
  auto alloc = [&](size_t bytes) -> void* {
    void* p = base + off;
    off = (off + bytes + 255) & ~(size_t)255;
    return p;
  };
  const size_t SD = (size_t)SEQ * DIMN;
  const size_t WW = (size_t)DIMN * DIMN;
  __bf16* xb  = (__bf16*)alloc(SD * 2);
  __bf16* wqb = (__bf16*)alloc(WW * 2);
  __bf16* wkb = (__bf16*)alloc(WW * 2);
  __bf16* wvb = (__bf16*)alloc(WW * 2);
  __bf16* wob = (__bf16*)alloc(WW * 2);
  float*  pf  = (float*) alloc(SD * 4);   // f32 projection scratch (reused)
  __bf16* qb  = (__bf16*)alloc(SD * 2);
  __bf16* kb  = (__bf16*)alloc(SD * 2);
  __bf16* vT  = (__bf16*)alloc(SD * 2);
  __bf16* ob  = (__bf16*)alloc(SD * 2);

  const int TPB = 256;
  // 1) casts to bf16
  cast_f32_bf16<<<(int)(SD / TPB), TPB, 0, stream>>>(x,  xb,  (int)SD);
  cast_f32_bf16<<<(int)(WW / TPB), TPB, 0, stream>>>(Wq, wqb, (int)WW);
  cast_f32_bf16<<<(int)(WW / TPB), TPB, 0, stream>>>(Wk, wkb, (int)WW);
  cast_f32_bf16<<<(int)(WW / TPB), TPB, 0, stream>>>(Wv, wvb, (int)WW);
  cast_f32_bf16<<<(int)(WW / TPB), TPB, 0, stream>>>(Wo, wob, (int)WW);

  dim3 ggrid(DIMN / 256, SEQ / 128);  // (8, 24)
  const float qscale = 0.088388347648318447f;  // 1/sqrt(128)

  // 2) Q projection -> rms+rope (scale folded into q)
  gemm_bf16_nt<<<ggrid, 256, 0, stream>>>(xb, wqb, bq, pf, SEQ, DIMN, DIMN);
  rmsnorm_rope<<<SEQ, 256, 0, stream>>>(pf, gq, freqs, qb, qscale);
  // 3) K projection -> rms+rope
  gemm_bf16_nt<<<ggrid, 256, 0, stream>>>(xb, wkb, bk, pf, SEQ, DIMN, DIMN);
  rmsnorm_rope<<<SEQ, 256, 0, stream>>>(pf, gk, freqs, kb, 1.0f);
  // 4) V projection -> transpose-cast to [DIMN, SEQ]
  gemm_bf16_nt<<<ggrid, 256, 0, stream>>>(xb, wvb, bv, pf, SEQ, DIMN, DIMN);
  transpose_cast_bf16<<<(int)(SD / TPB), TPB, 0, stream>>>(pf, vT, SEQ, DIMN);
  // 5) attention
  dim3 agrid(SEQ / 64, NHEAD);  // (48, 16), 4 waves/block
  flash_attn<<<agrid, 128, 0, stream>>>(qb, kb, vT, ob);
  // 6) output projection straight into d_out (f32)
  gemm_bf16_nt<<<ggrid, 256, 0, stream>>>(ob, wob, bo, outp, SEQ, DIMN, DIMN);
}
